// GMM_13838384628109
// MI455X (gfx1250) — matrix operand order repaired and checked by reading, not compiled, not run on hardware
//
#include <hip/hip_runtime.h>

#define N_ROWS 100000
#define D_DIM  512
#define K_CL   32
#define H1_DIM 512
#define H2_DIM 256
#define MROWS  64
#define SROWS  64
#define SITERS 16
#define SBLK_ROWS (SROWS * SITERS)
#define LN2PI_2F 0.91893853320467274178f

typedef __attribute__((ext_vector_type(16))) __bf16 v16bf;
typedef __attribute__((ext_vector_type(8)))  float  v8f;
typedef __attribute__((ext_vector_type(4)))  unsigned int v4u;

union V16U { v16bf v; v4u q[2]; };

__device__ __forceinline__ unsigned short f2bf16(float f) {
  union { float f; unsigned u; } x; x.f = f;
  unsigned r = x.u + 0x7FFFu + ((x.u >> 16) & 1u);   // round-to-nearest-even
  return (unsigned short)(r >> 16);
}

__device__ __forceinline__ v8f v8zero() {
  v8f z = {0.f, 0.f, 0.f, 0.f, 0.f, 0.f, 0.f, 0.f};
  return z;
}

__device__ __forceinline__ v8f wmma_bf16(v16bf a, v16bf b, v8f c) {
  return __builtin_amdgcn_wmma_f32_16x16x32_bf16(false, a, false, b, (short)0, c,
                                                 false, false);
}

// A operand: 16x32 (MxK) tile from row-major bf16 (stride rs elements).
__device__ __forceinline__ v16bf load_a(const unsigned short* base, int rs,
                                        int r0, int k0, int lane) {
  const unsigned short* p =
      base + (size_t)(r0 + (lane & 15)) * rs + k0 + ((lane >> 4) << 3);
  V16U u;
  u.q[0] = *(const v4u*)p;
  u.q[1] = *(const v4u*)(p + 16);
  return u.v;
}

// B operand: 32x16 (KxN) tile read from transposed storage Bt[col][k] (stride rs).
__device__ __forceinline__ v16bf load_b(const unsigned short* base, int rs,
                                        int c0, int k0, int lane) {
  const unsigned short* p =
      base + (size_t)(c0 + (lane & 15)) * rs + k0 + ((lane >> 4) << 3);
  V16U u;
  u.q[0] = *(const v4u*)p;
  u.q[1] = *(const v4u*)(p + 16);
  return u.v;
}

__device__ __forceinline__ float softplus_f(float v) {
  return fmaxf(v, 0.f) + __logf(1.f + __expf(-fabsf(v)));
}

// --- gfx1250 async global->LDS copy (ASYNCcnt path, ISA 15.18.3 op 98) -------
__device__ __forceinline__ unsigned lds_off_u32(const void* p) {
  return (unsigned)(unsigned long long)(const __attribute__((address_space(3))) char*)p;
}

__device__ __forceinline__ void async_g2l_b128(unsigned lds_off, const void* gptr) {
  asm volatile("global_load_async_to_lds_b128 %0, %1, off"
               :: "v"(lds_off), "v"((unsigned long long)(size_t)gptr)
               : "memory");
}

__device__ __forceinline__ void wait_async0() {
  asm volatile("s_wait_asynccnt 0x0" ::: "memory");
}

// ---------------------------------------------------------------- prep kernels
__global__ void k_convT(const float* __restrict__ W, unsigned short* __restrict__ Wt,
                        int rows, int cols) {
  int i = blockIdx.x * 256 + threadIdx.x;
  if (i < rows * cols) {
    int r = i / cols, c = i % cols;
    Wt[c * rows + r] = f2bf16(W[i]);        // Wt[col][k] = W[k][col]
  }
}

__global__ void k_zero(float* __restrict__ p, int n) {
  int i = blockIdx.x * 256 + threadIdx.x;
  if (i < n) p[i] = 0.f;
}

// ------------------------------------------------- kernel 1: MLP + softmax att
struct MlpShared {                       // overlap xbf with (h2bf, logits)
  union {
    unsigned short xbf[MROWS][D_DIM + 16];
    struct {
      unsigned short h2bf[MROWS][H2_DIM + 16];
      float logits[MROWS][K_CL];
    } s2;
  };
};

__global__ __launch_bounds__(256) void k_mlp_att(
    const float* __restrict__ x,
    const unsigned short* __restrict__ W1t, const float* __restrict__ b1,
    const unsigned short* __restrict__ W2t, const float* __restrict__ b2,
    const unsigned short* __restrict__ W3t, const float* __restrict__ b3,
    float* __restrict__ att, float* __restrict__ att_sum) {
  __shared__ __attribute__((aligned(32))) MlpShared U;
  __shared__ __attribute__((aligned(32))) unsigned short h1bf[MROWS][H1_DIM + 16];

  const int tid = threadIdx.x;
  const int wave = tid >> 5, lane = tid & 31;
  const long row0 = (long)blockIdx.x * MROWS;

  // prime L2/WGP$ with this wave's first weight panel
  __builtin_prefetch(W1t + (size_t)(((wave >> 2) << 4) + (lane & 15)) * D_DIM, 0, 0);

  // stage x tile as bf16 (zero-padded past N)
  for (int i = tid; i < MROWS * (D_DIM / 4); i += 256) {
    int r = i >> 7, c4 = i & 127;
    long gr = row0 + r;
    float4 v = make_float4(0.f, 0.f, 0.f, 0.f);
    if (gr < N_ROWS) v = *(const float4*)(x + gr * D_DIM + c4 * 4);
    int c = c4 * 4;
    U.xbf[r][c]     = f2bf16(v.x);
    U.xbf[r][c + 1] = f2bf16(v.y);
    U.xbf[r][c + 2] = f2bf16(v.z);
    U.xbf[r][c + 3] = f2bf16(v.w);
  }
  __syncthreads();

  const int rt = (wave & 3) << 4;
  const int rbase = rt + ((lane >> 4) << 3);

  // ---- layer 1: h1 = softplus(x @ W1 + b1)
  for (int ct = (wave >> 2); ct < H1_DIM / 16; ct += 2) {
    v8f acc = v8zero();
#pragma unroll 4
    for (int k0 = 0; k0 < D_DIM; k0 += 32) {
      v16bf a = load_a(&U.xbf[0][0], D_DIM + 16, rt, k0, lane);
      v16bf b = load_b(W1t, D_DIM, ct << 4, k0, lane);
      acc = wmma_bf16(a, b, acc);
    }
    int col = (ct << 4) + (lane & 15);
    float bias = b1[col];
#pragma unroll
    for (int r = 0; r < 8; ++r)
      h1bf[rbase + r][col] = f2bf16(softplus_f(acc[r] + bias));
  }
  __syncthreads();   // xbf dead from here; LDS re-used as h2bf/logits

  // ---- layer 2: h2 = softplus(h1 @ W2 + b2)
  for (int ct = (wave >> 2); ct < H2_DIM / 16; ct += 2) {
    v8f acc = v8zero();
#pragma unroll 4
    for (int k0 = 0; k0 < H1_DIM; k0 += 32) {
      v16bf a = load_a(&h1bf[0][0], H1_DIM + 16, rt, k0, lane);
      v16bf b = load_b(W2t, H1_DIM, ct << 4, k0, lane);
      acc = wmma_bf16(a, b, acc);
    }
    int col = (ct << 4) + (lane & 15);
    float bias = b2[col];
#pragma unroll
    for (int r = 0; r < 8; ++r)
      U.s2.h2bf[rbase + r][col] = f2bf16(softplus_f(acc[r] + bias));
  }
  __syncthreads();

  // ---- layer 3: logits = h2 @ W3 + b3  (2 col-tiles; one per wave half)
  {
    int ct = wave >> 2;
    v8f acc = v8zero();
#pragma unroll
    for (int k0 = 0; k0 < H2_DIM; k0 += 32) {
      v16bf a = load_a(&U.s2.h2bf[0][0], H2_DIM + 16, rt, k0, lane);
      v16bf b = load_b(W3t, H2_DIM, ct << 4, k0, lane);
      acc = wmma_bf16(a, b, acc);
    }
    int col = (ct << 4) + (lane & 15);
    float bias = b3[col];
#pragma unroll
    for (int r = 0; r < 8; ++r)
      U.s2.logits[rbase + r][col] = acc[r] + bias;
  }
  __syncthreads();

  // ---- softmax over K=32 per row; write att; accumulate att_sum
  if (tid < MROWS) {
    long gr = row0 + tid;
    float l[K_CL];
    float mx = -1e30f;
#pragma unroll
    for (int k = 0; k < K_CL; ++k) { l[k] = U.s2.logits[tid][k]; mx = fmaxf(mx, l[k]); }
    float s = 0.f;
#pragma unroll
    for (int k = 0; k < K_CL; ++k) { l[k] = __expf(l[k] - mx); s += l[k]; }
    float inv = 1.f / s;
    bool valid = gr < N_ROWS;
#pragma unroll
    for (int k = 0; k < K_CL; ++k) {
      float a = valid ? l[k] * inv : 0.f;
      U.s2.logits[tid][k] = a;
      if (valid) att[gr * K_CL + k] = a;
    }
  }
  __syncthreads();

  if (tid < K_CL) {
    float s = 0.f;
    for (int r = 0; r < MROWS; ++r) s += U.s2.logits[r][tid];
    atomicAdd(&att_sum[tid], s);
  }
}

// ------------------------------------ kernel 2: att.T @ x and att.T @ (x*x)
__global__ __launch_bounds__(256) void k_stats(
    const float* __restrict__ x, const float* __restrict__ att,
    float* __restrict__ mean_num, float* __restrict__ ex2_num) {
  __shared__ __attribute__((aligned(32))) unsigned short xT[D_DIM][SROWS + 8];
  __shared__ __attribute__((aligned(32))) unsigned short aT[K_CL][SROWS + 8];

  const int tid = threadIdx.x;
  const int wave = tid >> 5, lane = tid & 31;
  const long base = (long)blockIdx.x * SBLK_ROWS;

  const int kt = (wave & 1) << 4;    // k row-tile (output M)
  const int d0 = (wave >> 1) << 7;   // 128-wide d quarter (output N)

  v8f accM[8], accE[8];
#pragma unroll
  for (int t = 0; t < 8; ++t) { accM[t] = v8zero(); accE[t] = v8zero(); }

  for (int it = 0; it < SITERS; ++it) {
    long r0 = base + (long)it * SROWS;
    __syncthreads();     // previous iteration's WMMA reads done
    // stage x, K(n)-contiguous (transposed) for the B operand
    for (int i = tid; i < SROWS * (D_DIM / 4); i += 256) {
      int r = i >> 7, c4 = i & 127;
      long gr = r0 + r;
      float4 v = make_float4(0.f, 0.f, 0.f, 0.f);
      if (gr < N_ROWS) v = *(const float4*)(x + gr * D_DIM + c4 * 4);
      int c = c4 * 4;
      xT[c][r]     = f2bf16(v.x);
      xT[c + 1][r] = f2bf16(v.y);
      xT[c + 2][r] = f2bf16(v.z);
      xT[c + 3][r] = f2bf16(v.w);
    }
    // stage att transposed: aT[k][n]
    for (int i = tid; i < SROWS * K_CL; i += 256) {
      int r = i >> 5, k = i & 31;
      long gr = r0 + r;
      float a = (gr < N_ROWS) ? att[gr * K_CL + k] : 0.f;
      aT[k][r] = f2bf16(a);
    }
    __syncthreads();

#pragma unroll
    for (int n0 = 0; n0 < SROWS; n0 += 32) {
      v16bf a = load_a(&aT[0][0], SROWS + 8, kt, n0, lane);
#pragma unroll
      for (int t = 0; t < 8; ++t) {
        v16bf bx = load_b(&xT[0][0], SROWS + 8, d0 + (t << 4), n0, lane);
        accM[t] = wmma_bf16(a, bx, accM[t]);
        v16bf b2 = bx * bx;               // v_pk_mul_bf16, co-executes with XDL
        accE[t] = wmma_bf16(a, b2, accE[t]);
      }
    }
  }

  // commit partial [32,512] accumulators
  const int kb = kt + ((lane >> 4) << 3);
#pragma unroll
  for (int t = 0; t < 8; ++t) {
    int d = d0 + (t << 4) + (lane & 15);
#pragma unroll
    for (int r = 0; r < 8; ++r) {
      atomicAdd(&mean_num[(kb + r) * D_DIM + d], accM[t][r]);
      atomicAdd(&ex2_num[(kb + r) * D_DIM + d], accE[t][r]);
    }
  }
}

// ------------------------------------------- kernel 3: means/vars + B panels
__global__ __launch_bounds__(256) void k_finalize(
    const float* __restrict__ mean_num, const float* __restrict__ ex2_num,
    const float* __restrict__ att_sum, float* __restrict__ vars_out,
    unsigned short* __restrict__ iv_bf, unsigned short* __restrict__ miv_bf,
    float* __restrict__ ck, float* __restrict__ logdet) {
  __shared__ float sred[2][256];
  int k = blockIdx.x, tid = threadIdx.x;
  float inv_s = 1.f / att_sum[k];
  float lck = 0.f, lld = 0.f;
  for (int d = tid; d < D_DIM; d += 256) {
    float m  = mean_num[k * D_DIM + d] * inv_s;
    float e2 = ex2_num[k * D_DIM + d] * inv_s;
    float v  = e2 - m * m;
    vars_out[k * D_DIM + d] = v;
    float iv = 1.f / v;
    iv_bf[k * D_DIM + d]  = f2bf16(iv);
    miv_bf[k * D_DIM + d] = f2bf16(m * iv);
    lck += m * m * iv;
    lld += __logf(v);
  }
  sred[0][tid] = lck; sred[1][tid] = lld;
  __syncthreads();
  for (int s = 128; s > 0; s >>= 1) {
    if (tid < s) { sred[0][tid] += sred[0][tid + s]; sred[1][tid] += sred[1][tid + s]; }
    __syncthreads();
  }
  if (tid == 0) { ck[k] = sred[0][0]; logdet[k] = sred[1][0]; }
}

// ------------------------- kernel 4: fused Mahalanobis GEMMs + log-likelihood
__global__ __launch_bounds__(256) void k_gmm(
    const float* __restrict__ x,
    const unsigned short* __restrict__ iv_bf, const unsigned short* __restrict__ miv_bf,
    const float* __restrict__ ck, const float* __restrict__ logdet,
    float* __restrict__ out) {
  __shared__ __attribute__((aligned(32))) unsigned short xbf[MROWS][D_DIM + 16];
  __shared__ __attribute__((aligned(32))) unsigned short lds_iv[K_CL][D_DIM + 16];
  __shared__ __attribute__((aligned(32))) unsigned short lds_miv[K_CL][D_DIM + 16];

  const int tid = threadIdx.x;
  const int wave = tid >> 5, lane = tid & 31;
  const long row0 = (long)blockIdx.x * MROWS;

  // async-stage the two 32x512 bf16 B panels into padded LDS (ASYNCcnt path)
  {
    const unsigned iv0  = lds_off_u32(&lds_iv[0][0]);
    const unsigned miv0 = lds_off_u32(&lds_miv[0][0]);
    for (int e = tid * 8; e < K_CL * D_DIM; e += 256 * 8) {
      int row = e >> 9;                   // e / D_DIM
      int col = e & (D_DIM - 1);
      unsigned loff = (unsigned)((row * (D_DIM + 16) + col) * 2);
      async_g2l_b128(iv0 + loff,  iv_bf  + e);
      async_g2l_b128(miv0 + loff, miv_bf + e);
    }
  }

  // stage x tile as bf16 (regular path, needs f32->bf16 conversion)
  for (int i = tid; i < MROWS * (D_DIM / 4); i += 256) {
    int r = i >> 7, c4 = i & 127;
    long gr = row0 + r;
    float4 v = make_float4(0.f, 0.f, 0.f, 0.f);
    if (gr < N_ROWS) v = *(const float4*)(x + gr * D_DIM + c4 * 4);
    int c = c4 * 4;
    xbf[r][c]     = f2bf16(v.x);
    xbf[r][c + 1] = f2bf16(v.y);
    xbf[r][c + 2] = f2bf16(v.z);
    xbf[r][c + 3] = f2bf16(v.w);
  }
  wait_async0();
  __syncthreads();

  const int rt = (wave & 3) << 4;
  const int ct = wave >> 2;          // 0..1 (K=32 -> two 16-col tiles)
  v8f a1 = v8zero(), a2 = v8zero();
#pragma unroll 4
  for (int k0 = 0; k0 < D_DIM; k0 += 32) {
    v16bf ax   = load_a(&xbf[0][0], D_DIM + 16, rt, k0, lane);
    v16bf ax2  = ax * ax;            // x^2 operand via packed bf16 mul
    v16bf biv  = load_b(&lds_iv[0][0],  D_DIM + 16, ct << 4, k0, lane);
    v16bf bmiv = load_b(&lds_miv[0][0], D_DIM + 16, ct << 4, k0, lane);
    a1 = wmma_bf16(ax2, biv, a1);
    a2 = wmma_bf16(ax, bmiv, a2);
  }
  int col = (ct << 4) + (lane & 15);
  float cterm = -0.5f * ck[col] - 0.5f * logdet[col] - (float)D_DIM * LN2PI_2F;
  int rbase = rt + ((lane >> 4) << 3);
#pragma unroll
  for (int r = 0; r < 8; ++r) {
    long gr = row0 + rbase + r;
    if (gr < N_ROWS) out[gr * K_CL + col] = -0.5f * a1[r] + a2[r] + cterm;
  }
}

// ----------------------------------------------------------------- launcher
extern "C" void kernel_launch(void* const* d_in, const int* in_sizes, int n_in,
                              void* d_out, int out_size, void* d_ws, size_t ws_size,
                              hipStream_t stream) {
  (void)in_sizes; (void)n_in; (void)out_size; (void)ws_size;
  const float* x  = (const float*)d_in[0];
  const float* W1 = (const float*)d_in[1];
  const float* b1 = (const float*)d_in[2];
  const float* W2 = (const float*)d_in[3];
  const float* b2 = (const float*)d_in[4];
  const float* W3 = (const float*)d_in[5];
  const float* b3 = (const float*)d_in[6];

  float* gmm_out  = (float*)d_out;                              // [N, K]
  float* vars_out = (float*)d_out + (size_t)N_ROWS * K_CL;      // [K, D]

  char* ws = (char*)d_ws;
  size_t off = 0;
  auto alloc = [&](size_t bytes) -> void* {
    void* p = ws + off;
    off += (bytes + 255) & ~(size_t)255;
    return p;
  };
  unsigned short* W1t    = (unsigned short*)alloc((size_t)D_DIM * H1_DIM * 2);
  unsigned short* W2t    = (unsigned short*)alloc((size_t)H1_DIM * H2_DIM * 2);
  unsigned short* W3t    = (unsigned short*)alloc((size_t)H2_DIM * K_CL * 2);
  float* att             = (float*)alloc((size_t)N_ROWS * K_CL * 4);
  float* att_sum         = (float*)alloc(K_CL * 4);
  float* mean_num        = (float*)alloc((size_t)K_CL * D_DIM * 4);
  float* ex2_num         = (float*)alloc((size_t)K_CL * D_DIM * 4);
  unsigned short* iv_bf  = (unsigned short*)alloc((size_t)K_CL * D_DIM * 2);
  unsigned short* miv_bf = (unsigned short*)alloc((size_t)K_CL * D_DIM * 2);
  float* ck              = (float*)alloc(K_CL * 4);
  float* logdet          = (float*)alloc(K_CL * 4);

  k_convT<<<(D_DIM * H1_DIM + 255) / 256, 256, 0, stream>>>(W1, W1t, D_DIM, H1_DIM);
  k_convT<<<(H1_DIM * H2_DIM + 255) / 256, 256, 0, stream>>>(W2, W2t, H1_DIM, H2_DIM);
  k_convT<<<(H2_DIM * K_CL + 255) / 256, 256, 0, stream>>>(W3, W3t, H2_DIM, K_CL);
  k_zero<<<1, 256, 0, stream>>>(att_sum, K_CL);
  k_zero<<<(K_CL * D_DIM + 255) / 256, 256, 0, stream>>>(mean_num, K_CL * D_DIM);
  k_zero<<<(K_CL * D_DIM + 255) / 256, 256, 0, stream>>>(ex2_num, K_CL * D_DIM);

  int row_blocks = (N_ROWS + MROWS - 1) / MROWS;          // 1563
  k_mlp_att<<<row_blocks, 256, 0, stream>>>(x, W1t, b1, W2t, b2, W3t, b3, att, att_sum);

  int stat_blocks = (N_ROWS + SBLK_ROWS - 1) / SBLK_ROWS; // 98
  k_stats<<<stat_blocks, 256, 0, stream>>>(x, att, mean_num, ex2_num);

  k_finalize<<<K_CL, 256, 0, stream>>>(mean_num, ex2_num, att_sum, vars_out,
                                       iv_bf, miv_bf, ck, logdet);

  k_gmm<<<row_blocks, 256, 0, stream>>>(x, iv_bf, miv_bf, ck, logdet, gmm_out);
}
